// KNNSegmentator_26508538151473
// MI455X (gfx1250) — matrix operand order repaired and checked by reading, not compiled, not run on hardware
//
#include <hip/hip_runtime.h>
#include <hip/hip_bf16.h>

// ---------------------------------------------------------------------------
// KNN segmentator for MI455X (gfx1250, wave32, WMMA).
//   test_feature  : [8, 196, 384]  f32   (d_in[0])
//   train_features: [384, 50000]   f32   (d_in[1])
//   train_labels  : [256, 50000]   i32   (d_in[2])
//   out           : [8, 224, 224]  (pred class id, written as float)
// ---------------------------------------------------------------------------

typedef __attribute__((ext_vector_type(8)))  float          v8f;
typedef __attribute__((ext_vector_type(8)))  __bf16         v8bf;
typedef __attribute__((ext_vector_type(16))) __bf16         v16bf;
typedef __attribute__((ext_vector_type(8)))  unsigned short u16x8;
typedef __attribute__((ext_vector_type(16))) unsigned short u16x16;

namespace knncfg {
constexpr int K_NB   = 20;
constexpr int C_CLS  = 21;
constexpr int NROWSG = 14;
constexpr int HIMG   = 224;
constexpr int BS     = 8;
constexpr int NPATCH = 196;
constexpr int D      = 384;
constexpr int NTR    = 50000;
constexpr int MROWS  = BS * NPATCH;     // 1568
constexpr int MTILES = MROWS / 16;      // 98
constexpr int NSPLIT = 16;
constexpr int SLICE  = NTR / NSPLIT;    // 3125
constexpr int NCHUNK = 128;             // columns per inner chunk
constexpr int KBLK   = 64;              // k per LDS B block (2 wmma k-steps)
constexpr int ASTR   = 392;             // 384 + 8 pad (bf16 elems)
constexpr int BSTR   = 72;              // 64 + 8 pad  (bf16 elems)
constexpr int SIMSTR = 132;             // 128 + 4 pad (f32 elems)
constexpr int NCAND  = NSPLIT * K_NB;   // 320 merge candidates
}

// ---------------- bf16 split helpers (round-to-nearest-even) ----------------
__device__ __forceinline__ unsigned short f2bf(float f) {
  unsigned u = __float_as_uint(f);
  unsigned r = u + 0x7FFFu + ((u >> 16) & 1u);
  return (unsigned short)(r >> 16);
}
__device__ __forceinline__ float bf2f(unsigned short h) {
  return __uint_as_float(((unsigned)h) << 16);
}
__device__ __forceinline__ v16bf frag_cat(u16x8 a, u16x8 b) {
  u16x16 t = __builtin_shufflevector(a, b, 0, 1, 2, 3, 4, 5, 6, 7,
                                     8, 9, 10, 11, 12, 13, 14, 15);
  return __builtin_bit_cast(v16bf, t);
}
// A fragment: 16 contiguous bf16 (per-lane row, k = 16*half + s)
__device__ __forceinline__ v16bf ldfragA(const unsigned short* p) {
  u16x8 a = *(const u16x8*)p;
  u16x8 b = *(const u16x8*)(p + 8);
  return frag_cat(a, b);
}
// B fragment: 16 contiguous bf16 (per-lane column, k = 16*half + s)
__device__ __forceinline__ v16bf ldfragB(const unsigned short* p) {
  u16x8 a = *(const u16x8*)p;
  u16x8 b = *(const u16x8*)(p + 8);
  return frag_cat(a, b);
}

// ===========================================================================
// Kernel B: transpose + pack labels  [256][50000] i32 -> [50000][256] u8
// ===========================================================================
__global__ __launch_bounds__(256) void knn_label_pack(
    const int* __restrict__ labels, unsigned char* __restrict__ labT) {
  using namespace knncfg;
  __shared__ unsigned char tile[64 * 260];
  const int t  = threadIdx.x;
  const int n0 = blockIdx.x * 64;
#pragma unroll 4
  for (int rep = 0; rep < 64; ++rep) {      // 256 rows x 64 cols, coalesced
    int e = rep * 256 + t;
    int l = e >> 6;
    int j = e & 63;
    int n = n0 + j;
    int v = (n < NTR) ? labels[l * NTR + n] : 0;
    tile[j * 260 + l] = (unsigned char)v;
  }
  __syncthreads();
  unsigned* outw = (unsigned*)labT;
#pragma unroll 4
  for (int rep = 0; rep < 16; ++rep) {      // 64 rows x 64 dwords out
    int e  = rep * 256 + t;
    int jr = e >> 6;
    int dw = e & 63;
    int n  = n0 + jr;
    if (n < NTR) {
      const unsigned char* s = &tile[jr * 260 + dw * 4];
      unsigned v = (unsigned)s[0] | ((unsigned)s[1] << 8) |
                   ((unsigned)s[2] << 16) | ((unsigned)s[3] << 24);
      outw[n * 64 + dw] = v;
    }
  }
}

// ===========================================================================
// Kernel A: fused GEMM (bf16 hi/lo split WMMA, f32 accumulate) + top-20
// grid = MTILES * NSPLIT blocks of 256 threads (8 waves)
// ===========================================================================
__global__ __launch_bounds__(256) void knn_gemm_topk(
    const float* __restrict__ testf, const float* __restrict__ trainf,
    float* __restrict__ ws_vals, int* __restrict__ ws_idx) {
  using namespace knncfg;
  __shared__ unsigned short Ah[16 * ASTR];
  __shared__ unsigned short Al[16 * ASTR];
  __shared__ unsigned short Bth[NCHUNK * BSTR];
  __shared__ unsigned short Btl[NCHUNK * BSTR];
  __shared__ float          simL[16 * SIMSTR];
  __shared__ float          tv[16 * K_NB];
  __shared__ int            ti[16 * K_NB];

  const int t     = threadIdx.x;
  const int tile  = blockIdx.x / NSPLIT;
  const int slice = blockIdx.x % NSPLIT;
  const int lane  = t & 31;
  const int wave  = t >> 5;
  const int lhalf = lane >> 4;
  const int l15   = lane & 15;
  const int nstart = slice * SLICE;
  const int nend   = nstart + SLICE;

  // ---- stage A rows (16 x 384) as bf16 hi/lo into LDS ----
#pragma unroll
  for (int rep = 0; rep < 24; ++rep) {
    int e  = rep * 256 + t;         // 0..6143
    int lr = e / D;
    int k  = e - lr * D;
    float f = testf[(tile * 16 + lr) * D + k];
    unsigned short h = f2bf(f);
    unsigned short l = f2bf(f - bf2f(h));
    Ah[lr * ASTR + k] = h;
    Al[lr * ASTR + k] = l;
  }
  for (int i = t; i < 16 * K_NB; i += 256) { tv[i] = -3.4e38f; ti[i] = 0; }
  float minv = -3.4e38f;            // threshold cache, meaningful for t < 16

  for (int n0 = nstart; n0 < nend; n0 += NCHUNK) {
    v8f acc = {0.f, 0.f, 0.f, 0.f, 0.f, 0.f, 0.f, 0.f};

    for (int kb = 0; kb < D / KBLK; ++kb) {
      __syncthreads();              // prev fragments consumed / prev scan done
      // ---- stage B block [64 k][128 n] transposed as bf16 hi/lo ----
      const int k0 = kb * KBLK;
#pragma unroll 4
      for (int rep = 0; rep < 32; ++rep) {
        int e  = rep * 256 + t;     // 0..8191
        int kk = e >> 7;            // 0..63
        int j  = e & 127;
        int n  = n0 + j;
        float f = (n < NTR) ? trainf[(k0 + kk) * NTR + n] : 0.f;
        unsigned short h = f2bf(f);
        unsigned short l = f2bf(f - bf2f(h));
        Bth[j * BSTR + kk] = h;
        Btl[j * BSTR + kk] = l;
      }
      __syncthreads();
      // ---- 2 k-steps of 32, 3 WMMAs each (ah*bh + ah*bl + al*bh) ----
#pragma unroll
      for (int ks = 0; ks < KBLK; ks += 32) {
        int abase = l15 * ASTR + (k0 + ks) + lhalf * 16;
        v16bf afh = ldfragA(Ah + abase);
        v16bf afl = ldfragA(Al + abase);
        int bbase = (wave * 16 + l15) * BSTR + ks + lhalf * 16;
        v16bf bfh = ldfragB(Bth + bbase);
        v16bf bfl = ldfragB(Btl + bbase);
        acc = __builtin_amdgcn_wmma_f32_16x16x32_bf16(
            false, afh, false, bfh, (short)0, acc, false, false);
        acc = __builtin_amdgcn_wmma_f32_16x16x32_bf16(
            false, afh, false, bfl, (short)0, acc, false, false);
        acc = __builtin_amdgcn_wmma_f32_16x16x32_bf16(
            false, afl, false, bfh, (short)0, acc, false, false);
      }
    }

    // ---- dump 16x16 tile: VGPR v -> row lhalf*8+v, col wave*16+l15 ----
    {
      int col   = wave * 16 + l15;
      int rbase = lhalf * 8;
#pragma unroll
      for (int v = 0; v < 8; ++v) simL[(rbase + v) * SIMSTR + col] = acc[v];
    }
    __syncthreads();

    // ---- per-row top-20 maintenance (thread r owns row r) ----
    if (t < 16) {
      const float* srow  = simL + t * SIMSTR;
      float*       trow  = tv + t * K_NB;
      int*         tirow = ti + t * K_NB;
      int jmax = nend - n0;
      if (jmax > NCHUNK) jmax = NCHUNK;
      for (int j = 0; j < jmax; ++j) {
        float v = srow[j];
        if (v > minv) {
          int   ms = 0;
          float mv = trow[0];
#pragma unroll
          for (int s = 1; s < K_NB; ++s) {
            float x = trow[s];
            if (x < mv) { mv = x; ms = s; }
          }
          trow[ms]  = v;
          tirow[ms] = n0 + j;
          mv = trow[0];
#pragma unroll
          for (int s = 1; s < K_NB; ++s) mv = fminf(mv, trow[s]);
          minv = mv;
        }
      }
    }
  }

  __syncthreads();
  for (int i = t; i < 16 * K_NB; i += 256) {
    int rr  = i / K_NB;
    int j   = i - rr * K_NB;
    int off = ((tile * NSPLIT + slice) * 16 + rr) * K_NB + j;
    ws_vals[off] = tv[i];
    ws_idx[off]  = ti[i];
  }
}

// ===========================================================================
// Kernel C: merge 16 partial top-20 lists, softmax, gather labels, vote,
// argmax, scatter into make_grid pixel layout.  grid = MROWS blocks (1/patch)
// ===========================================================================
__global__ __launch_bounds__(256) void knn_merge_vote(
    const float* __restrict__ ws_vals, const int* __restrict__ ws_idx,
    const unsigned char* __restrict__ labT, float* __restrict__ out) {
  using namespace knncfg;
  __shared__ float cv[NCAND];
  __shared__ int   ci[NCAND];
  __shared__ float sv[K_NB];
  __shared__ int   si[K_NB];
  __shared__ float w[K_NB];

  const int t    = threadIdx.x;
  const int m    = blockIdx.x;
  const int tile = m >> 4;
  const int r    = m & 15;

  for (int i = t; i < NCAND; i += 256) {
    int s   = i / K_NB;
    int j   = i - s * K_NB;
    int off = ((tile * NSPLIT + s) * 16 + r) * K_NB + j;
    cv[i] = ws_vals[off];
    ci[i] = ws_idx[off];
  }
  __syncthreads();

  // rank-based top-20 selection (ties broken by candidate index)
  for (int i = t; i < NCAND; i += 256) {
    float v    = cv[i];
    int   rank = 0;
    for (int j = 0; j < NCAND; ++j) {
      float vj = cv[j];
      rank += (vj > v) || (vj == v && j < i);
    }
    if (rank < K_NB) { sv[rank] = v; si[rank] = ci[i]; }
  }
  __syncthreads();

  if (t == 0) {
    float mx = sv[0];
#pragma unroll
    for (int k = 1; k < K_NB; ++k) mx = fmaxf(mx, sv[k]);
    float s = 0.f;
#pragma unroll
    for (int k = 0; k < K_NB; ++k) {
      float e = __expf(sv[k] - mx);   // TEMP == 1
      w[k] = e;
      s += e;
    }
    float inv = 1.f / s;
#pragma unroll
    for (int k = 0; k < K_NB; ++k) w[k] *= inv;
  }
  __syncthreads();

  // one thread per pixel of the 16x16 patch
  float wk[K_NB];
  int   lab[K_NB];
#pragma unroll
  for (int k = 0; k < K_NB; ++k) {
    lab[k] = labT[si[k] * 256 + t];
    wk[k]  = w[k];
  }
  float bestv = -1.f;
  int   bestc = 0;
#pragma unroll
  for (int c = 0; c < C_CLS; ++c) {
    float v = 0.f;
#pragma unroll
    for (int k = 0; k < K_NB; ++k) v += (lab[k] == c) ? wk[k] : 0.f;
    if (v > bestv) { bestv = v; bestc = c; }
  }

  const int b  = m / NPATCH;
  const int p  = m - b * NPATCH;
  const int rg = p / NROWSG;
  const int cg = p - rg * NROWSG;
  const int py = t >> 4;
  const int px = t & 15;
  out[(b * HIMG + rg * 16 + py) * HIMG + cg * 16 + px] = (float)bestc;
}

// ===========================================================================
extern "C" void kernel_launch(void* const* d_in, const int* in_sizes, int n_in,
                              void* d_out, int out_size, void* d_ws,
                              size_t ws_size, hipStream_t stream) {
  using namespace knncfg;
  (void)in_sizes; (void)n_in; (void)out_size; (void)ws_size;

  const float* testf  = (const float*)d_in[0];
  const float* trainf = (const float*)d_in[1];
  const int*   labels = (const int*)d_in[2];
  float*       out    = (float*)d_out;

  // workspace layout
  unsigned char* ws      = (unsigned char*)d_ws;
  unsigned char* labT    = ws;                              // 12,800,000 B
  size_t         off1    = (size_t)NTR * 256;
  float*         ws_vals = (float*)(ws + off1);             // 2,007,040 B
  size_t         off2    = off1 + (size_t)MTILES * NSPLIT * 16 * K_NB * 4;
  int*           ws_idx  = (int*)(ws + off2);               // 2,007,040 B

  knn_label_pack<<<(NTR + 63) / 64, 256, 0, stream>>>(labels, labT);
  knn_gemm_topk<<<MTILES * NSPLIT, 256, 0, stream>>>(testf, trainf, ws_vals,
                                                     ws_idx);
  knn_merge_vote<<<MROWS, 256, 0, stream>>>(ws_vals, ws_idx, labT, out);
}
